// TrittentionCube_51384988729458
// MI455X (gfx1250) — matrix operand order repaired and checked by reading, not compiled, not run on hardware
//
#include <hip/hip_runtime.h>
#include <hip/hip_bf16.h>

typedef __bf16 bf16;
typedef __attribute__((ext_vector_type(16))) __bf16 v16bf;
typedef __attribute__((ext_vector_type(8)))  __bf16 v8bf;
typedef __attribute__((ext_vector_type(8)))  float  v8f;

#define IGNORE_LOGIT (-1.0e6f)
#define DUMMY_LOGIT  (-1000.0f)

__device__ __forceinline__ bf16 tobf(float x) { return (bf16)x; }

// A-fragment (16x32 bf16, M x K): lane m = lane&15,
// K half kl = (lane>=16)?16:0 ; 16 contiguous elements [kl,kl+16).
__device__ __forceinline__ v16bf load_a_frag(const bf16* A, int lda, int lane) {
  int m  = lane & 15;
  int kl = (lane >> 4) << 4;
  const bf16* p = A + m * lda + kl;
  return *(const v16bf*)(p);
}

// B-fragment (32x16 bf16, K x N) loaded from B^T stored row-major [N][K]:
// lane column = lane&15, K half kb = (lane>=16)?16:0, 16 contiguous values.
__device__ __forceinline__ v16bf load_b_frag(const bf16* Bt, int ldb, int lane) {
  int ncol = lane & 15;
  int kb = (lane >> 4) << 4;
  return *(const v16bf*)(Bt + ncol * ldb + kb);
}

__device__ __forceinline__ v8f wmma_bf16(v16bf a, v16bf b, v8f c) {
  return __builtin_amdgcn_wmma_f32_16x16x32_bf16(false, a, false, b, (short)0, c,
                                                 false, false);
}

// ---------------- prep kernels (f32 -> bf16, with transposes) ----------------

__global__ void k_cvt(const float* __restrict__ src, bf16* __restrict__ dst, int n) {
  int i = blockIdx.x * 256 + threadIdx.x;
  if (i < n) dst[i] = tobf(src[i]);
}

// W[n][d][h] -> WpT[(n*64+h)][d]   (proj weight, NT layout)
__global__ void k_prep_wproj(const float* __restrict__ W, bf16* __restrict__ dst) {
  int i = blockIdx.x * 256 + threadIdx.x;           // over 8*512*64
  int h = i & 63, d = (i >> 6) & 511, n = i >> 15;
  dst[(size_t)((n << 6) + h) * 512 + d] = tobf(W[i]);
}

// W_Vq[n][i][j][h] -> WvqT[n][(j*64+h)][i]
__global__ void k_prep_wvq(const float* __restrict__ W, bf16* __restrict__ dst) {
  int i = blockIdx.x * 256 + threadIdx.x;           // over 8*64*64*64
  int h = i & 63, j = (i >> 6) & 63, ii = (i >> 12) & 63, n = i >> 18;
  dst[(((size_t)n << 12) + (j << 6) + h) * 64 + ii] = tobf(W[i]);
}

// W_O[n][h][d] -> WoT[d][(n*64+h)]
__global__ void k_prep_wo(const float* __restrict__ W, bf16* __restrict__ dst) {
  int i = blockIdx.x * 256 + threadIdx.x;           // over 8*64*512
  int d = i & 511, h = (i >> 9) & 63, n = i >> 15;
  dst[(size_t)d * 512 + (n << 6) + h] = tobf(W[i]);
}

// ---------------- projection GEMM: [256,512] x [512,512]^T -> 5 proj bufs ----

__global__ __launch_bounds__(256) void k_gemm_proj(
    const bf16* __restrict__ A, const bf16* __restrict__ Bt,
    const float* __restrict__ bias, bf16* __restrict__ out) {
  int wave = threadIdx.x >> 5, lane = threadIdx.x & 31;
  int tile = blockIdx.x * 8 + wave;                 // 512 tiles (16 x 32)
  int row0 = (tile >> 5) << 4;
  int col0 = (tile & 31) << 4;
  v8f c = {};
  for (int k0 = 0; k0 < 512; k0 += 32) {
    v16bf a = load_a_frag(A + row0 * 512 + k0, 512, lane);
    v16bf b = load_b_frag(Bt + col0 * 512 + k0, 512, lane);
    c = wmma_bf16(a, b, c);
  }
  int col = col0 + (lane & 15);
  int n = col >> 6, h = col & 63;
  float bv = bias[col];
  int mb = row0 + ((lane >> 4) << 3);
  for (int e = 0; e < 8; ++e)
    out[(size_t)((n << 8) + mb + e) * 64 + h] = tobf(c[e] + bv);
}

// ---- per-head GEMM [256,64] x [4096,64]^T -> [256,4096] (step1) -------------

__global__ __launch_bounds__(256) void k_gemm_head(
    const bf16* __restrict__ Aall, const bf16* __restrict__ Btall,
    bf16* __restrict__ outall) {
  int wave = threadIdx.x >> 5, lane = threadIdx.x & 31;
  int gt = blockIdx.x * 8 + wave;                   // 32768 tiles total
  int n = gt >> 12;
  int row0 = ((gt >> 8) & 15) << 4;
  int col0 = (gt & 255) << 4;
  const bf16* A  = Aall + ((size_t)n << 14);
  const bf16* Bt = Btall + ((size_t)n << 18);
  v8f c = {};
  for (int k0 = 0; k0 < 64; k0 += 32) {
    v16bf a = load_a_frag(A + row0 * 64 + k0, 64, lane);
    v16bf b = load_b_frag(Bt + col0 * 64 + k0, 64, lane);
    c = wmma_bf16(a, b, c);
  }
  bf16* out = outall + ((size_t)n << 20);
  int col = col0 + (lane & 15);
  int mb = row0 + ((lane >> 4) << 3);
  for (int e = 0; e < 8; ++e)
    out[(size_t)(mb + e) * 4096 + col] = tobf(c[e]);
}

// ---- per-head GEMM with transposed epilogue (tmp): out [n][h][s][j] ---------
// tmp_t[n][h][s][j] = sum_i v1[n][s][i] * W_Vq[n][i][j][h]

__global__ __launch_bounds__(256) void k_gemm_head_t(
    const bf16* __restrict__ Aall, const bf16* __restrict__ Btall,
    bf16* __restrict__ outall) {
  int wave = threadIdx.x >> 5, lane = threadIdx.x & 31;
  int gt = blockIdx.x * 8 + wave;                   // 32768 tiles total
  int n = gt >> 12;
  int row0 = ((gt >> 8) & 15) << 4;
  int col0 = (gt & 255) << 4;
  const bf16* A  = Aall + ((size_t)n << 14);
  const bf16* Bt = Btall + ((size_t)n << 18);
  v8f c = {};
  for (int k0 = 0; k0 < 64; k0 += 32) {
    v16bf a = load_a_frag(A + row0 * 64 + k0, 64, lane);
    v16bf b = load_b_frag(Bt + col0 * 64 + k0, 64, lane);
    c = wmma_bf16(a, b, c);
  }
  int col = col0 + (lane & 15);                     // col = j*64 + h
  int h = col & 63, j = col >> 6;
  int mb = row0 + ((lane >> 4) << 3);               // s base
  bf16* out = outall + ((size_t)((n << 6) + h) << 14) + j;
  for (int e = 0; e < 8; ++e)
    out[(size_t)(mb + e) << 6] = tobf(c[e]);
}

// ---------------- fused trittention core: one workgroup per (n, r) -----------

__global__ __launch_bounds__(256) void k_mega(
    const bf16* __restrict__ k1a, const bf16* __restrict__ k2a,
    const bf16* __restrict__ v2a, const bf16* __restrict__ s1a,
    const bf16* __restrict__ tmpa, bf16* __restrict__ zb) {
  __shared__ alignas(32) bf16 S[256 * 256];          // 128 KB scores -> P
  __shared__ alignas(32) unsigned char regB[256 * 64 * 4];  // s2t (bf16) then Amat (f32)
  __shared__ alignas(32) bf16 v2T[64 * 256];         // 32 KB  v2^T [j][t]
  __shared__ float red[256];

  bf16*  s2t  = (bf16*)regB;
  float* Amat = (float*)regB;

  const int tid  = threadIdx.x;
  const int wave = tid >> 5;
  const int lane = tid & 31;
  const int r = blockIdx.x;
  const int n = blockIdx.y;

  // init S to IGNORE; stage v2^T into LDS
  {
    bf16 ig = tobf(IGNORE_LOGIT);
    v8bf igv = { ig, ig, ig, ig, ig, ig, ig, ig };
    v8bf* Sv = (v8bf*)S;
    for (int i = tid; i < 8192; i += 256) Sv[i] = igv;
    const bf16* v2 = v2a + ((size_t)n << 14);
    for (int i = tid; i < 16384; i += 256) {
      int t = i >> 6, j = i & 63;
      v2T[j * 256 + t] = v2[i];
    }
  }

  // pass0: s2t[t][i] = sum_j step1[n,r][i][j] * k2[n][t][j]   (WMMA, store D^T)
  {
    const bf16* A  = s1a + ((size_t)((n << 8) + r) << 12);   // [64 i][64 j]
    const bf16* Bt = k2a + ((size_t)n << 14);                // [256 t][64 j]
    for (int tt = wave; tt < 64; tt += 8) {
      int i0 = (tt >> 4) << 4;
      int t0 = (tt & 15) << 4;
      v8f c = {};
      for (int k0 = 0; k0 < 64; k0 += 32) {
        v16bf a = load_a_frag(A + i0 * 64 + k0, 64, lane);
        v16bf b = load_b_frag(Bt + t0 * 64 + k0, 64, lane);
        c = wmma_bf16(a, b, c);
      }
      int mb = i0 + ((lane >> 4) << 3);
      int tc = t0 + (lane & 15);
      for (int e = 0; e < 8; ++e)
        s2t[tc * 64 + mb + e] = tobf(c[e]);
    }
  }
  __syncthreads();

  // pass1: S[s][t] = (s<t<r) ? (k1[s,:]·s2t[t,:])/64 : IGNORE   (WMMA)
  float mloc = IGNORE_LOGIT;
  {
    const bf16* A = k1a + ((size_t)n << 14);                 // [256 s][64 i]
    for (int tt = wave; tt < 256; tt += 8) {
      int s0 = (tt >> 4) << 4;
      int t0 = (tt & 15) << 4;
      int tmax = (t0 + 15 < r - 1) ? (t0 + 15) : (r - 1);
      if (!((t0 < r) && (s0 < tmax))) continue;              // tile fully masked
      v8f c = {};
      for (int k0 = 0; k0 < 64; k0 += 32) {
        v16bf a = load_a_frag(A + s0 * 64 + k0, 64, lane);
        v16bf b = load_b_frag(s2t + t0 * 64 + k0, 64, lane);
        c = wmma_bf16(a, b, c);
      }
      int mb = s0 + ((lane >> 4) << 3);
      int t = t0 + (lane & 15);
      for (int e = 0; e < 8; ++e) {
        int s = mb + e;
        float val = (s < t && t < r) ? c[e] * (1.0f / 64.0f) : IGNORE_LOGIT;
        S[s * 256 + t] = tobf(val);
        mloc = fmaxf(mloc, val);
      }
    }
  }
  red[tid] = mloc;
  __syncthreads();
  for (int off = 128; off > 0; off >>= 1) {
    if (tid < off) red[tid] = fmaxf(red[tid], red[tid + off]);
    __syncthreads();
  }
  const float mval = fmaxf(red[0], DUMMY_LOGIT);   // dummy logit joins the max
  __syncthreads();

  // pass2: unnormalized softmax numerators in-place; accumulate Z.
  // Rows s >= r contain only IGNORE (P would be 0) -> skip them entirely.
  float lsum = 0.0f;
  {
    const int lim = r << 8;                        // r * 256 entries
    for (int i = tid; i < lim; i += 256) {
      float xv = (float)S[i];
      float ev = __expf(xv - mval);
      S[i] = tobf(ev);
      lsum += ev;
    }
  }
  red[tid] = lsum;
  __syncthreads();
  for (int off = 128; off > 0; off >>= 1) {
    if (tid < off) red[tid] += red[tid + off];
    __syncthreads();
  }
  const float Zinv = 1.0f / (red[0] + __expf(DUMMY_LOGIT - mval));
  __syncthreads();   // regB reuse boundary: s2t dead, Amat live

  // pass3: Amat[s][j] = sum_t P[s][t] * v2[t][j]   (WMMA)
  // P rows s >= r-1 are zero -> skip those s-tiles; P cols t >= r are zero
  // (exponentiated rows only) -> clamp the K loop to ceil(r/32)*32.
  {
    const int kmax = ((r + 31) >> 5) << 5;         // multiple of 32, <= 256+31
    const int kend = kmax < 256 ? kmax : 256;
    for (int tt = wave; tt < 64; tt += 8) {
      int s0 = (tt >> 2) << 4;
      int j0 = (tt & 3) << 4;
      if (s0 >= r - 1) continue;                   // Amat rows never read
      v8f c = {};
      for (int k0 = 0; k0 < kend; k0 += 32) {
        v16bf a = load_a_frag(S + s0 * 256 + k0, 256, lane);
        v16bf b = load_b_frag(v2T + j0 * 256 + k0, 256, lane);
        c = wmma_bf16(a, b, c);
      }
      int mb = s0 + ((lane >> 4) << 3);
      int jc = j0 + (lane & 15);
      for (int e = 0; e < 8; ++e)
        Amat[(mb + e) * 64 + jc] = c[e];
    }
  }
  __syncthreads();

  // pass4: z[h] = Zinv * sum_{s<r-1, j} Amat[s][j] * tmp_t[n][h][s][j]
  // tmp_t is [n][h][s][j]: contiguous (s,j) per h -> b128 vector loads.
  {
    int h = tid & 63;
    int g = tid >> 6;
    float acc = 0.0f;
    const bf16* base = tmpa + ((size_t)((n << 6) + h) << 14);
    int send = (g << 6) + 64;
    if (send > r - 1) send = r - 1;
    for (int s = (g << 6); s < send; ++s) {
      const float* Ar = Amat + s * 64;
      const bf16* row = base + ((size_t)s << 6);
      for (int j0 = 0; j0 < 64; j0 += 8) {
        v8bf tv = *(const v8bf*)(row + j0);
        for (int e = 0; e < 8; ++e)
          acc += Ar[j0 + e] * (float)tv[e];
      }
    }
    red[tid] = acc;
  }
  __syncthreads();
  if (tid < 64) {
    float zf = (red[tid] + red[tid + 64] + red[tid + 128] + red[tid + 192]) * Zinv;
    zb[r * 512 + (n << 6) + tid] = tobf(zf);
  }
}

// ---------------- output GEMM: zb[256,512] x WoT[512,512]^T + b_O -> f32 -----

__global__ __launch_bounds__(256) void k_gemm_out(
    const bf16* __restrict__ A, const bf16* __restrict__ Bt,
    const float* __restrict__ bO, float* __restrict__ out) {
  int wave = threadIdx.x >> 5, lane = threadIdx.x & 31;
  int tile = blockIdx.x * 8 + wave;
  int row0 = (tile >> 5) << 4;
  int col0 = (tile & 31) << 4;
  v8f c = {};
  for (int k0 = 0; k0 < 512; k0 += 32) {
    v16bf a = load_a_frag(A + row0 * 512 + k0, 512, lane);
    v16bf b = load_b_frag(Bt + col0 * 512 + k0, 512, lane);
    c = wmma_bf16(a, b, c);
  }
  int col = col0 + (lane & 15);
  float bv = bO[col];
  int mb = row0 + ((lane >> 4) << 3);
  for (int e = 0; e < 8; ++e)
    out[(size_t)(mb + e) * 512 + col] = c[e] + bv;
}

// ---------------- workspace layout (bytes) ----------------

static constexpr size_t OFF_XB    = 0;                       // 256*512 bf16
static constexpr size_t OFF_WPT   = OFF_XB    + 262144;      // 5*512*512 bf16
static constexpr size_t OFF_PROJ  = OFF_WPT   + 2621440;     // 5*8*256*64 bf16
static constexpr size_t OFF_WKQ   = OFF_PROJ  + 1310720;     // 8*4096*64 bf16
static constexpr size_t OFF_WVQT  = OFF_WKQ   + 4194304;     // 8*4096*64 bf16
static constexpr size_t OFF_WOT   = OFF_WVQT  + 4194304;     // 512*512 bf16
static constexpr size_t OFF_STEP1 = OFF_WOT   + 524288;      // 8*256*4096 bf16
static constexpr size_t OFF_TMP   = OFF_STEP1 + 16777216;    // 8*64*256*64 bf16
static constexpr size_t OFF_ZB    = OFF_TMP   + 16777216;    // 256*512 bf16

extern "C" void kernel_launch(void* const* d_in, const int* in_sizes, int n_in,
                              void* d_out, int out_size, void* d_ws, size_t ws_size,
                              hipStream_t stream) {
  const float* x   = (const float*)d_in[0];
  const float* Wp[5] = {(const float*)d_in[1], (const float*)d_in[2],
                        (const float*)d_in[3], (const float*)d_in[4],
                        (const float*)d_in[5]};
  const float* Wkq = (const float*)d_in[6];
  const float* Wvq = (const float*)d_in[7];
  const float* Wo  = (const float*)d_in[8];
  const float* bp[5] = {(const float*)d_in[9], (const float*)d_in[10],
                        (const float*)d_in[11], (const float*)d_in[12],
                        (const float*)d_in[13]};
  const float* bO  = (const float*)d_in[14];

  char* ws = (char*)d_ws;
  bf16* xb    = (bf16*)(ws + OFF_XB);
  bf16* WpT   = (bf16*)(ws + OFF_WPT);
  bf16* proj  = (bf16*)(ws + OFF_PROJ);
  bf16* Wkqb  = (bf16*)(ws + OFF_WKQ);
  bf16* WvqT  = (bf16*)(ws + OFF_WVQT);
  bf16* WoT   = (bf16*)(ws + OFF_WOT);
  bf16* step1 = (bf16*)(ws + OFF_STEP1);
  bf16* tmp   = (bf16*)(ws + OFF_TMP);
  bf16* zb    = (bf16*)(ws + OFF_ZB);

  k_cvt<<<512, 256, 0, stream>>>(x, xb, 131072);
  for (int p = 0; p < 5; ++p)
    k_prep_wproj<<<1024, 256, 0, stream>>>(Wp[p], WpT + (size_t)p * 262144);
  k_cvt<<<8192, 256, 0, stream>>>(Wkq, Wkqb, 2097152);
  k_prep_wvq<<<8192, 256, 0, stream>>>(Wvq, WvqT);
  k_prep_wo<<<1024, 256, 0, stream>>>(Wo, WoT);

  for (int p = 0; p < 5; ++p)
    k_gemm_proj<<<64, 256, 0, stream>>>(xb, WpT + (size_t)p * 262144, bp[p],
                                        proj + (size_t)p * 131072);

  const bf16* k1 = proj + (size_t)0 * 131072;
  const bf16* k2 = proj + (size_t)1 * 131072;
  const bf16* q  = proj + (size_t)2 * 131072;
  const bf16* v1 = proj + (size_t)3 * 131072;
  const bf16* v2 = proj + (size_t)4 * 131072;

  k_gemm_head<<<4096, 256, 0, stream>>>(q, Wkqb, step1);     // step1[n][r][i*64+j]
  k_gemm_head_t<<<4096, 256, 0, stream>>>(v1, WvqT, tmp);    // tmp_t[n][h][s][j]

  k_mega<<<dim3(256, 8), 256, 0, stream>>>(k1, k2, v2, step1, tmp, zb);

  k_gemm_out<<<64, 256, 0, stream>>>(zb, WoT, bO, (float*)d_out);
}
